// HSMM_generative_4243427689002
// MI455X (gfx1250) — matrix operand order repaired and checked by reading, not compiled, not run on hardware
//
#include <hip/hip_runtime.h>
#include <hip/hip_bf16.h>
#include <math.h>

#define NEGINF (-1.0e38f)

typedef __attribute__((ext_vector_type(16))) __bf16 v16bf;
typedef __attribute__((ext_vector_type(8)))  float  v8f;

union Frag { uint4 q[2]; v16bf v; };

__device__ __forceinline__ unsigned short f2bf(float x) {
    unsigned u = __builtin_bit_cast(unsigned, x);
    return (unsigned short)((u + 0x7FFFu + ((u >> 16) & 1u)) >> 16);  // RNE
}

__device__ __forceinline__ unsigned pack_bf2(float lo, float hi) {
    return (unsigned)f2bf(lo) | ((unsigned)f2bf(hi) << 16);
}

// LDS byte-offset of a generic pointer into __shared__ (addrspace(3) low bits)
__device__ __forceinline__ unsigned lds_off(void* p) {
    return (unsigned)(unsigned long long)(__attribute__((address_space(3))) void*)p;
}

// GLOBAL_LOAD_ASYNC_TO_LDS_B128: DMA 16B/lane global -> LDS, ASYNCcnt-tracked.
__device__ __forceinline__ void async_copy_b128(unsigned lds_byte_off, const void* gptr) {
    asm volatile("global_load_async_to_lds_b128 %0, %1, off"
                 :: "v"(lds_byte_off), "v"(gptr) : "memory");
}

__device__ __forceinline__ void wait_async0() {
#if __has_builtin(__builtin_amdgcn_s_wait_asynccnt)
    __builtin_amdgcn_s_wait_asynccnt(0);
#else
    asm volatile("s_wait_asynccnt 0x0" ::: "memory");
#endif
}

// ---------------------------------------------------------------------------
// Elementwise f32 -> packed bf16 (n must be a multiple of 4).
// ---------------------------------------------------------------------------
__global__ __launch_bounds__(256) void pack_bf16_kernel(
    const float* __restrict__ X, unsigned short* __restrict__ Y, int n4)
{
    const int i = blockIdx.x * 256 + threadIdx.x;
    if (i < n4) {
        float4 f = ((const float4*)X)[i];
        uint2 o;
        o.x = pack_bf2(f.x, f.y);
        o.y = pack_bf2(f.z, f.w);
        ((uint2*)Y)[i] = o;
    }
}

// ---------------------------------------------------------------------------
// W[K][N] f32 -> WT[N][K] bf16 (tiled 64x64 through LDS). K,N multiples of 64.
// ---------------------------------------------------------------------------
__global__ __launch_bounds__(256) void transpose_pack_bf16(
    const float* __restrict__ W, unsigned short* __restrict__ WT, int K, int N)
{
    __shared__ float tile[64][65];
    const int tid = threadIdx.x;
    const int k0 = blockIdx.y << 6;
    const int n0 = blockIdx.x << 6;
    {
        const int r  = tid >> 2;          // 0..63 (k within tile)
        const int cs = (tid & 3) << 4;    // 0,16,32,48 (n within tile)
        const float* src = W + (size_t)(k0 + r) * N + n0 + cs;
        float4 f0 = ((const float4*)src)[0];
        float4 f1 = ((const float4*)src)[1];
        float4 f2 = ((const float4*)src)[2];
        float4 f3 = ((const float4*)src)[3];
        float* d = &tile[r][cs];
        d[0]=f0.x; d[1]=f0.y; d[2]=f0.z; d[3]=f0.w;
        d[4]=f1.x; d[5]=f1.y; d[6]=f1.z; d[7]=f1.w;
        d[8]=f2.x; d[9]=f2.y; d[10]=f2.z; d[11]=f2.w;
        d[12]=f3.x; d[13]=f3.y; d[14]=f3.z; d[15]=f3.w;
    }
    __syncthreads();
    {
        const int rn   = tid >> 2;        // 0..63 (n within tile)
        const int kseg = (tid & 3) << 4;  // 0,16,32,48 (k within tile)
        uint4 o0, o1;
        o0.x = pack_bf2(tile[kseg+ 0][rn], tile[kseg+ 1][rn]);
        o0.y = pack_bf2(tile[kseg+ 2][rn], tile[kseg+ 3][rn]);
        o0.z = pack_bf2(tile[kseg+ 4][rn], tile[kseg+ 5][rn]);
        o0.w = pack_bf2(tile[kseg+ 6][rn], tile[kseg+ 7][rn]);
        o1.x = pack_bf2(tile[kseg+ 8][rn], tile[kseg+ 9][rn]);
        o1.y = pack_bf2(tile[kseg+10][rn], tile[kseg+11][rn]);
        o1.z = pack_bf2(tile[kseg+12][rn], tile[kseg+13][rn]);
        o1.w = pack_bf2(tile[kseg+14][rn], tile[kseg+15][rn]);
        uint4* dst = (uint4*)(WT + (size_t)(n0 + rn) * K + k0 + kseg);
        dst[0] = o0;
        dst[1] = o1;
    }
}

// ---------------------------------------------------------------------------
// bf16 WMMA GEMM, async-DMA staged + double-buffered.
//   C[M,N] = act(A[M,K] @ BT[N,K]^T + bias[N])
// A, BT are bf16 row-major with contiguous K, so tiles stream into LDS via
// global_load_async_to_lds_b128 with zero VALU work. Block tile 128x128,
// 256 threads = 8 waves (4 M-strips x 2 N-strips), wave = 2x4 WMMA tiles.
// LDS rows padded to 40 bf16 -> conflict-free ds_load_b128 fragment reads.
// mode 0: f32 out (+bias). mode 1: bf16 out (+bias, relu).
// ---------------------------------------------------------------------------
#define LDSTR 40

__global__ __launch_bounds__(256) void gemm_bf16_async(
    const unsigned short* __restrict__ A,   // M x K bf16
    const unsigned short* __restrict__ BT,  // N x K bf16
    const float* __restrict__ bias,
    void* __restrict__ C,
    int M, int N, int Kdim, int mode)
{
    (void)M;
    __shared__ unsigned short Asl[2][128 * LDSTR];
    __shared__ unsigned short Bsl[2][128 * LDSTR];

    const int tid  = threadIdx.x;
    const int wid  = tid >> 5;
    const int lane = tid & 31;
    const int lm   = lane & 15;
    const int lh   = lane >> 4;
    const int waveM = wid & 3;
    const int waveN = wid >> 2;
    const int bm = blockIdx.y << 7;
    const int bn = blockIdx.x << 7;

    v8f zero = {0.f,0.f,0.f,0.f,0.f,0.f,0.f,0.f};
    v8f acc[2][4];
    #pragma unroll
    for (int ms = 0; ms < 2; ++ms)
        #pragma unroll
        for (int ns = 0; ns < 4; ++ns) acc[ms][ns] = zero;

    // Each tile = 128 rows x 64B; 2 issues/thread cover it (16B per lane).
    auto copy_tiles = [&](int bufi, int k0) {
        #pragma unroll
        for (int i = 0; i < 2; ++i) {
            const int li  = (i << 8) + tid;       // 0..511
            const int row = li >> 2;
            const int sb  = (li & 3) << 4;        // byte segment 0/16/32/48
            const char* ga = (const char*)A  + (((size_t)(bm + row) * Kdim + k0) << 1) + sb;
            async_copy_b128(lds_off(&Asl[bufi][row * LDSTR]) + sb, ga);
            const char* gb = (const char*)BT + (((size_t)(bn + row) * Kdim + k0) << 1) + sb;
            async_copy_b128(lds_off(&Bsl[bufi][row * LDSTR]) + sb, gb);
        }
    };

    copy_tiles(0, 0);
    const int nk = Kdim >> 5;
    for (int ki = 0; ki < nk; ++ki) {
        const int cur = ki & 1;
        wait_async0();        // my DMAs into buf[cur] done
        __syncthreads();      // everyone's DMAs done; prior reads of buf[cur^1] done
        if (ki + 1 < nk) copy_tiles(cur ^ 1, (ki + 1) << 5);

        Frag afr[2];
        #pragma unroll
        for (int ms = 0; ms < 2; ++ms) {
            const int row = (waveM << 5) + (ms << 4) + lm;
            const unsigned short* pA = &Asl[cur][row * LDSTR + lh * 8];
            afr[ms].q[0] = *(const uint4*)pA;          // k = lh*8 + 0..7
            afr[ms].q[1] = *(const uint4*)(pA + 16);   // k = 16 + lh*8 + 0..7
        }
        #pragma unroll
        for (int ns = 0; ns < 4; ++ns) {
            const int col = (waveN << 6) + (ns << 4) + lm;
            const unsigned short* pB = &Bsl[cur][col * LDSTR + (lh << 3)];
            Frag bfr;
            bfr.q[0] = *(const uint4*)pB;              // k = lh*8 + 0..7
            bfr.q[1] = *(const uint4*)(pB + 16);       // k = 16 + lh*8 + 0..7
            #pragma unroll
            for (int ms = 0; ms < 2; ++ms)
                acc[ms][ns] = __builtin_amdgcn_wmma_f32_16x16x32_bf16(
                    false, afr[ms].v, false, bfr.v, (short)0, acc[ms][ns], false, false);
        }
    }

    #pragma unroll
    for (int ms = 0; ms < 2; ++ms) {
        #pragma unroll
        for (int ns = 0; ns < 4; ++ns) {
            const int col = bn + (waveN << 6) + (ns << 4) + lm;
            const float bv = bias[col];
            #pragma unroll
            for (int r = 0; r < 8; ++r) {
                const int row = bm + (waveM << 5) + (ms << 4) + (lh << 3) + r;
                float v = acc[ms][ns][r] + bv;
                if (mode == 1) {
                    v = fmaxf(v, 0.0f);
                    ((unsigned short*)C)[(size_t)row * N + col] = f2bf(v);
                } else {
                    ((float*)C)[(size_t)row * N + col] = v;
                }
            }
        }
    }
}

// ---------------------------------------------------------------------------
// tscores[k][j] = A_from[k,:] . A_to[:,j] + (k==j ? NEGINF : 0)
// ---------------------------------------------------------------------------
__global__ void tscores_kernel(const float* __restrict__ Af,
                               const float* __restrict__ At,
                               float* __restrict__ tsc)
{
    const int j = threadIdx.x;
    const int k = blockIdx.x;
    float acc = 0.f;
    #pragma unroll 8
    for (int a = 0; a < 64; ++a) acc += Af[k * 64 + a] * At[a * 128 + j];
    tsc[k * 128 + j] = acc + (j == k ? NEGINF : 0.0f);
}

// ---------------------------------------------------------------------------
// trans[b,k,j] = tsc[k,j] + sum_a from[b,k,a]*to[b,j,a]  (WMMA; B operand is
// K-contiguous in memory so fragments load straight from global).
// ---------------------------------------------------------------------------
__global__ __launch_bounds__(256) void einsum_trans_kernel(
    const float* __restrict__ cond, const float* __restrict__ tsc,
    float* __restrict__ trans)
{
    const int tid  = threadIdx.x;
    const int wid  = tid >> 5;
    const int lane = tid & 31;
    const int g  = blockIdx.x * 8 + wid;
    const int b  = g >> 3;
    const int mt = g & 7;
    const int lm = lane & 15;
    const int lh = lane >> 4;

    const float* base = cond + (size_t)b * 16384;
    v8f zero = {0.f,0.f,0.f,0.f,0.f,0.f,0.f,0.f};
    v8f acc[8];
    #pragma unroll
    for (int nt = 0; nt < 8; ++nt) acc[nt] = zero;

    #pragma unroll
    for (int ks = 0; ks < 64; ks += 32) {
        Frag fa;
        {
            const float* ap = base + ((mt << 4) + lm) * 128 + ks + lh * 8;
            float4 g0 = ((const float4*)ap)[0];
            float4 g1 = ((const float4*)ap)[1];
            float4 g2 = ((const float4*)(ap + 16))[0];
            float4 g3 = ((const float4*)(ap + 16))[1];
            fa.q[0].x = pack_bf2(g0.x, g0.y); fa.q[0].y = pack_bf2(g0.z, g0.w);
            fa.q[0].z = pack_bf2(g1.x, g1.y); fa.q[0].w = pack_bf2(g1.z, g1.w);
            fa.q[1].x = pack_bf2(g2.x, g2.y); fa.q[1].y = pack_bf2(g2.z, g2.w);
            fa.q[1].z = pack_bf2(g3.x, g3.y); fa.q[1].w = pack_bf2(g3.z, g3.w);
        }
        #pragma unroll
        for (int nt = 0; nt < 8; ++nt) {
            const float* bp = base + ((nt << 4) + lm) * 128 + 64 + ks + (lh << 3);
            float4 h0 = ((const float4*)bp)[0];
            float4 h1 = ((const float4*)bp)[1];
            float4 h2 = ((const float4*)bp)[4];
            float4 h3 = ((const float4*)bp)[5];
            Frag fb;
            fb.q[0].x = pack_bf2(h0.x, h0.y); fb.q[0].y = pack_bf2(h0.z, h0.w);
            fb.q[0].z = pack_bf2(h1.x, h1.y); fb.q[0].w = pack_bf2(h1.z, h1.w);
            fb.q[1].x = pack_bf2(h2.x, h2.y); fb.q[1].y = pack_bf2(h2.z, h2.w);
            fb.q[1].z = pack_bf2(h3.x, h3.y); fb.q[1].w = pack_bf2(h3.z, h3.w);
            acc[nt] = __builtin_amdgcn_wmma_f32_16x16x32_bf16(
                false, fa.v, false, fb.v, (short)0, acc[nt], false, false);
        }
    }
    #pragma unroll
    for (int nt = 0; nt < 8; ++nt) {
        const int col = (nt << 4) + lm;
        #pragma unroll
        for (int r = 0; r < 8; ++r) {
            const int row = (mt << 4) + (lh << 3) + r;
            trans[(size_t)b * 16384 + row * 128 + col] =
                acc[nt][r] + tsc[row * 128 + col];
        }
    }
}

// ---------------------------------------------------------------------------
// Row softmax over length-128 rows, one wave32 per row.
// mode 0: log_softmax (pi).  mode 1: softmax probabilities (P).
// ---------------------------------------------------------------------------
__global__ __launch_bounds__(256) void row_softmax_kernel(float* __restrict__ X,
                                                          int nrows, int mode)
{
    const int wid  = threadIdx.x >> 5;
    const int lane = threadIdx.x & 31;
    const int row  = blockIdx.x * 8 + wid;
    if (row >= nrows) return;
    float* p = X + (size_t)row * 128;
    float v[4];
    #pragma unroll
    for (int i = 0; i < 4; ++i) v[i] = p[lane + 32 * i];
    float m = fmaxf(fmaxf(v[0], v[1]), fmaxf(v[2], v[3]));
    #pragma unroll
    for (int off = 16; off > 0; off >>= 1) m = fmaxf(m, __shfl_xor(m, off, 32));
    float s = 0.f;
    #pragma unroll
    for (int i = 0; i < 4; ++i) s += __expf(v[i] - m);
    #pragma unroll
    for (int off = 16; off > 0; off >>= 1) s += __shfl_xor(s, off, 32);
    if (mode == 0) {
        const float l = m + __logf(s);
        #pragma unroll
        for (int i = 0; i < 4; ++i) p[lane + 32 * i] = v[i] - l;
    } else {
        const float inv = 1.0f / s;
        #pragma unroll
        for (int i = 0; i < 4; ++i) p[lane + 32 * i] = __expf(v[i] - m) * inv;
    }
}

// ---------------------------------------------------------------------------
// HSMM forward scan: one 128-thread WG per batch element, P[b] (64 KB)
// LDS-resident across all 60 steps; recurrence in probability space.
// ---------------------------------------------------------------------------
__global__ __launch_bounds__(128) void hsmm_scan_kernel(
    const float* __restrict__ P,
    const float* __restrict__ pi,
    const float* __restrict__ obs,   // (6, 60, 1024, 128)
    float* __restrict__ out)
{
    extern __shared__ float sm[];
    float* Pl  = sm;
    float* buf = sm + 16384;
    float* wv  = buf + 6 * 128;
    float* red = wv + 128;

    const int b = blockIdx.x;
    const int k = threadIdx.x;
    const float len_lp = -1.7917594909667969f;  // -log(6)

    for (int i = k; i < 16384; i += 128) Pl[i] = P[(size_t)b * 16384 + i];
    buf[k] = pi[b * 128 + k];
    #pragma unroll
    for (int l = 1; l < 6; ++l) buf[l * 128 + k] = NEGINF;
    __syncthreads();

    int head = 0;
    for (int t = 0; t < 60; ++t) {
        float vv[6];
        float m = -INFINITY;
        #pragma unroll
        for (int l = 0; l < 6; ++l) {
            if (t - l >= 0) {
                const float o = obs[(((size_t)l * 60 + (t - l)) * 1024 + b) * 128 + k];
                const float x = buf[((head + l) % 6) * 128 + k] + o;
                vv[l] = x;
                m = fmaxf(m, x);
            } else {
                vv[l] = NEGINF;
            }
        }
        float s = 0.f;
        #pragma unroll
        for (int l = 0; l < 6; ++l)
            if (t - l >= 0) s += __expf(vv[l] - m);
        const float alpha = m + __logf(s) + len_lp;

        red[k] = alpha;
        __syncthreads();
        for (int off = 64; off > 0; off >>= 1) {
            if (k < off) red[k] = fmaxf(red[k], red[k + off]);
            __syncthreads();
        }
        const float amax = red[0];
        __syncthreads();
        const float wk = __expf(alpha - amax);
        wv[k] = wk;
        __syncthreads();

        if (t == 59) {
            red[k] = wk;
            __syncthreads();
            for (int off = 64; off > 0; off >>= 1) {
                if (k < off) red[k] += red[k + off];
                __syncthreads();
            }
            if (k == 0) out[b] = amax + __logf(red[0]);
        } else {
            float acc = 0.f;
            #pragma unroll 8
            for (int kk = 0; kk < 128; ++kk) acc += wv[kk] * Pl[kk * 128 + k];
            const float astar = (acc > 0.f) ? (amax + __logf(acc)) : NEGINF;
            head = (head + 5) % 6;
            __syncthreads();
            buf[head * 128 + k] = astar;
            __syncthreads();
        }
    }
}

// ---------------------------------------------------------------------------
extern "C" void kernel_launch(void* const* d_in, const int* in_sizes, int n_in,
                              void* d_out, int out_size, void* d_ws, size_t ws_size,
                              hipStream_t stream) {
    (void)in_sizes; (void)n_in; (void)out_size; (void)ws_size;

    const float* uniqenc = (const float*)d_in[0];  // (1024, 1024)
    const float* obs     = (const float*)d_in[1];  // (6, 60, 1024, 128)
    const float* W_init  = (const float*)d_in[2];  // (1024, 128)
    const float* b_init  = (const float*)d_in[3];
    const float* A_from  = (const float*)d_in[4];  // (128, 64)
    const float* A_to    = (const float*)d_in[5];  // (64, 128)
    const float* W_c1    = (const float*)d_in[6];  // (1024, 8192)
    const float* b_c1    = (const float*)d_in[7];
    const float* W_c2    = (const float*)d_in[8];  // (8192, 16384)
    const float* b_c2    = (const float*)d_in[9];
    float* out = (float*)d_out;                    // (1024,)

    char* ws = (char*)d_ws;
    size_t off = 0;
    auto carve = [&](size_t bytes) { size_t o = off; off = (off + bytes + 255) & ~(size_t)255; return o; };
    float*          pi    = (float*)(ws + carve((size_t)1024 * 128 * 4));
    float*          tsc   = (float*)(ws + carve((size_t)128 * 128 * 4));
    unsigned short* ubf   = (unsigned short*)(ws + carve((size_t)1024 * 1024 * 2));
    unsigned short* WiT   = (unsigned short*)(ws + carve((size_t)128 * 1024 * 2));
    unsigned short* Wc1T  = (unsigned short*)(ws + carve((size_t)8192 * 1024 * 2));
    unsigned short* hbf   = (unsigned short*)(ws + carve((size_t)1024 * 8192 * 2));
    size_t wc2t_off       = carve((size_t)16384 * 8192 * 2);
    unsigned short* Wc2T  = (unsigned short*)(ws + wc2t_off);
    float*          cond  = (float*)(ws + carve((size_t)1024 * 16384 * 4));
    // trans aliases Wc2T: Wc2T is dead after GEMM3, trans is written after it.
    float*          trans = (float*)(ws + wc2t_off);

    // 0) precision/layout prep: bf16 activations, bf16 pre-transposed weights
    pack_bf16_kernel<<<1024, 256, 0, stream>>>(uniqenc, ubf, 1024 * 1024 / 4);
    transpose_pack_bf16<<<dim3(128 / 64, 1024 / 64), 256, 0, stream>>>(W_init, WiT, 1024, 128);
    transpose_pack_bf16<<<dim3(8192 / 64, 1024 / 64), 256, 0, stream>>>(W_c1, Wc1T, 1024, 8192);
    transpose_pack_bf16<<<dim3(16384 / 64, 8192 / 64), 256, 0, stream>>>(W_c2, Wc2T, 8192, 16384);

    // 1) pi_raw = uniqenc @ W_init + b_init   (f32 out)
    gemm_bf16_async<<<dim3(1, 8), 256, 0, stream>>>(ubf, WiT, b_init, pi, 1024, 128, 1024, 0);
    // 2) h = relu(uniqenc @ W_c1 + b_c1)      (bf16 out)
    gemm_bf16_async<<<dim3(64, 8), 256, 0, stream>>>(ubf, Wc1T, b_c1, hbf, 1024, 8192, 1024, 1);
    // 3) cond = h @ W_c2 + b_c2               (f32 out)
    gemm_bf16_async<<<dim3(128, 8), 256, 0, stream>>>(hbf, Wc2T, b_c2, cond, 1024, 16384, 8192, 0);
    // 4) tscores = A_from @ A_to + diag(NEGINF)
    tscores_kernel<<<128, 128, 0, stream>>>(A_from, A_to, tsc);
    // 5) trans_raw = tscores + einsum('bka,bja->bkj', ...)
    einsum_trans_kernel<<<1024, 256, 0, stream>>>(cond, tsc, trans);
    // 6) trans -> softmax probabilities P (in place)
    row_softmax_kernel<<<16384, 256, 0, stream>>>(trans, 1024 * 128, 1);
    // 7) pi -> log_softmax (in place)
    row_softmax_kernel<<<128, 256, 0, stream>>>(pi, 1024, 0);
    // 8) HSMM scan -> out
    const size_t shm = (16384 + 6 * 128 + 128 + 128) * sizeof(float);  // 69632 B
    hsmm_scan_kernel<<<1024, 128, shm, stream>>>(trans, pi, obs, out);
}
